// SupervisedGraphSage_75204877353221
// MI455X (gfx1250) — compile-verified, matrix operand order checked by reading, not run on hardware
//
#include <hip/hip_runtime.h>

// ---------------------------------------------------------------------------
// GraphSAGE forward for gfx1250 (MI455X): gather+mean staged to LDS, GEMMs via
// v_wmma_f32_16x16x32_f16 (wave32). Memory-bound on the feature gather
// (~1.2 GB logical, mostly L2-resident); WMMA work is ~6.5 GFLOP.
// ---------------------------------------------------------------------------

typedef __attribute__((ext_vector_type(16))) _Float16 v16h;
typedef __attribute__((ext_vector_type(8)))  float    v8f;

struct U32x8 { uint4 a, b; };

__device__ __forceinline__ unsigned short f2h(float f) {
  _Float16 h = (_Float16)f;                 // v_cvt_f16_f32 (RNE)
  return __builtin_bit_cast(unsigned short, h);
}

// A/B fragment element->K mapping for 16-bit 16x32 WMMA operands (ISA 7.12.2):
// lane l, element e (0..15): K = (e<8 ? (l>>4)*8 + e : 16 + (l>>4)*8 + (e-8))
__device__ __forceinline__ int koffs(int lane, int e) {
  int hi = (lane >> 4) * 8;
  return (e < 8) ? (hi + e) : (16 + hi + (e - 8));
}

// Load a 16x32 f16 A-fragment from an LDS row: K-chunk base kb already folded
// into p; second half of the fragment lives at K+16 (=> +16 u16 elements).
__device__ __forceinline__ v16h frag_strided(const unsigned short* p) {
  U32x8 s;
  s.a = *(const uint4*)(p);
  s.b = *(const uint4*)(p + 16);
  return __builtin_bit_cast(v16h, s);
}

// Load a pre-packed B-fragment (16 contiguous u16 per lane).
__device__ __forceinline__ v16h frag_packed(const unsigned short* p) {
  U32x8 s;
  s.a = *(const uint4*)(p);
  s.b = *(const uint4*)(p + 8);
  return __builtin_bit_cast(v16h, s);
}

// ---------------- weight packing kernels (f32 -> f16 B-fragments) -----------

// W1 [32,1024] -> w1pk[(ntile*32 + kc)*32 + lane][16]; ntile in {0,1}, kc 0..31
__global__ void pack_w1_kernel(const float* __restrict__ W1,
                               unsigned short* __restrict__ w1pk) {
  int t = blockIdx.x * blockDim.x + threadIdx.x;
  if (t >= 2 * 32 * 32 * 16) return;
  int e = t & 15, lane = (t >> 4) & 31, kc = (t >> 9) & 31, ntile = t >> 14;
  int n = ntile * 16 + (lane & 15);
  int k = kc * 32 + koffs(lane, e);
  w1pk[t] = f2h(W1[n * 1024 + k]);
}

// W2 [32,64] -> w2pk[(ntile*2 + kc)*32 + lane][16]; ntile in {0,1}, kc in {0,1}
__global__ void pack_w2_kernel(const float* __restrict__ W2,
                               unsigned short* __restrict__ w2pk) {
  int t = blockIdx.x * blockDim.x + threadIdx.x;
  if (t >= 2 * 2 * 32 * 16) return;
  int e = t & 15, lane = (t >> 4) & 31, kc = (t >> 9) & 1, ntile = t >> 10;
  int n = ntile * 16 + (lane & 15);
  int k = kc * 32 + koffs(lane, e);
  w2pk[t] = f2h(W2[n * 64 + k]);
}

// Wc [40,32] padded to N=48 -> wcpk[ntile*32 + lane][16]; ntile in {0,1,2}
__global__ void pack_wc_kernel(const float* __restrict__ Wc,
                               unsigned short* __restrict__ wcpk) {
  int t = blockIdx.x * blockDim.x + threadIdx.x;
  if (t >= 3 * 32 * 16) return;
  int e = t & 15, lane = (t >> 4) & 31, ntile = t >> 9;
  int n = ntile * 16 + (lane & 15);
  int k = koffs(lane, e);
  wcpk[t] = f2h((n < 40) ? Wc[n * 32 + k] : 0.0f);
}

// ---------------- layer 1: enc1 over 98304 rows -----------------------------
// rows 0..16383        -> nodes[]
// rows 16384..98303    -> neigh2 flattened
// h = relu([self_f | mean5(neigh_f)] @ W1^T)   -> h1 [98304, 32] f32
__global__ __launch_bounds__(256) void enc1_kernel(
    const float* __restrict__ feat, const int* __restrict__ nodes,
    const int* __restrict__ neigh1, const int* __restrict__ neigh2,
    const unsigned short* __restrict__ w1pk, float* __restrict__ h1) {
  // 16 rows x 1024 f16, row padded +8 u16 (=16B) to dodge LDS bank conflicts
  __shared__ unsigned short combS[16][1032];

  const int t = threadIdx.x;
  const int r = t >> 4;   // row in tile (0..15)
  const int j = t & 15;   // 16 threads cooperate on one row
  const long m = (long)blockIdx.x * 16 + r;

  const int idx = (m < 16384) ? nodes[m] : neigh2[m - 16384];
  const float* selfp = feat + (size_t)idx * 512;
  const int* nb = neigh1 + (size_t)idx * 5;
  const float* np0 = feat + (size_t)nb[0] * 512;
  const float* np1 = feat + (size_t)nb[1] * 512;
  const float* np2 = feat + (size_t)nb[2] * 512;
  const float* np3 = feat + (size_t)nb[3] * 512;
  const float* np4 = feat + (size_t)nb[4] * 512;

#pragma unroll
  for (int i = 0; i < 8; ++i) {
    const int c = j * 4 + i * 64;  // coalesced float4 across the 16 threads
    float4 s  = *(const float4*)(selfp + c);
    float4 a0 = *(const float4*)(np0 + c);
    float4 a1 = *(const float4*)(np1 + c);
    float4 a2 = *(const float4*)(np2 + c);
    float4 a3 = *(const float4*)(np3 + c);
    float4 a4 = *(const float4*)(np4 + c);
    float4 mn;
    mn.x = (a0.x + a1.x + a2.x + a3.x + a4.x) * 0.2f;
    mn.y = (a0.y + a1.y + a2.y + a3.y + a4.y) * 0.2f;
    mn.z = (a0.z + a1.z + a2.z + a3.z + a4.z) * 0.2f;
    mn.w = (a0.w + a1.w + a2.w + a3.w + a4.w) * 0.2f;
    *(ushort4*)&combS[r][c] =
        make_ushort4(f2h(s.x), f2h(s.y), f2h(s.z), f2h(s.w));
    *(ushort4*)&combS[r][512 + c] =
        make_ushort4(f2h(mn.x), f2h(mn.y), f2h(mn.z), f2h(mn.w));
  }
  __syncthreads();

  const int wave = t >> 5;
  if (wave < 2) {  // wave-uniform branch: EXEC all ones inside
    const int lane = t & 31;
    const int hi = lane >> 4;
    const unsigned short* arow = &combS[lane & 15][0];
    const unsigned short* bbase = w1pk + ((size_t)(wave * 32) * 32 + lane) * 16;
    v8f acc = {};
#pragma unroll
    for (int kc = 0; kc < 32; ++kc) {
      v16h A = frag_strided(arow + kc * 32 + hi * 8);
      v16h B = frag_packed(bbase + (size_t)kc * (32 * 16));
      acc = __builtin_amdgcn_wmma_f32_16x16x32_f16(
          false, A, false, B, (short)0, acc, false, false);
    }
    const long mbase = (long)blockIdx.x * 16 + hi * 8;
    const int ncol = wave * 16 + (lane & 15);
#pragma unroll
    for (int rr = 0; rr < 8; ++rr) {
      float v = acc[rr];
      h1[(mbase + rr) * 32 + ncol] = v > 0.0f ? v : 0.0f;
    }
  }
}

// ---------------- layer 2 + classifier --------------------------------------
// comb2 = [h1[b] | mean5(h1[16384 + b*5 + j])]  (16384 x 64)
// out   = relu(comb2 @ W2^T) @ Wc^T             (16384 x 40)
__global__ __launch_bounds__(256) void layer2_kernel(
    const float* __restrict__ h1, const unsigned short* __restrict__ w2pk,
    const unsigned short* __restrict__ wcpk, float* __restrict__ out) {
  __shared__ unsigned short comb2S[16][72];  // 64 + 8 pad (keeps 16B rows)
  __shared__ unsigned short h2S[16][40];     // 32 + 8 pad

  const int t = threadIdx.x;
  const int r = t >> 4, j = t & 15;
  const long m = (long)blockIdx.x * 16 + r;

#pragma unroll
  for (int q = 0; q < 4; ++q) {
    const int c = j * 4 + q;
    float v;
    if (c < 32) {
      v = h1[m * 32 + c];
    } else {
      const int hh = c - 32;
      float s = 0.0f;
#pragma unroll
      for (int jj = 0; jj < 5; ++jj)
        s += h1[(size_t)(16384 + m * 5 + jj) * 32 + hh];
      v = s * 0.2f;
    }
    comb2S[r][c] = f2h(v);
  }
  __syncthreads();

  const int wave = t >> 5, lane = t & 31, hi = lane >> 4;
  if (wave < 2) {  // hidden layer: 16x64 @ 64x16 per wave (2 K-chunks)
    v8f acc = {};
    const unsigned short* arow = &comb2S[lane & 15][0];
#pragma unroll
    for (int kc = 0; kc < 2; ++kc) {
      v16h A = frag_strided(arow + kc * 32 + hi * 8);
      v16h B = frag_packed(w2pk + ((size_t)(wave * 2 + kc) * 32 + lane) * 16);
      acc = __builtin_amdgcn_wmma_f32_16x16x32_f16(
          false, A, false, B, (short)0, acc, false, false);
    }
    const int ncol = wave * 16 + (lane & 15);
#pragma unroll
    for (int rr = 0; rr < 8; ++rr) {
      float v = acc[rr];
      h2S[rr + hi * 8][ncol] = f2h(v > 0.0f ? v : 0.0f);
    }
  }
  __syncthreads();

  if (wave < 3) {  // classifier: 16x32 @ 32x16 per wave, N padded 40 -> 48
    v16h A = frag_strided(&h2S[lane & 15][hi * 8]);
    v16h B = frag_packed(wcpk + ((size_t)(wave * 32) + lane) * 16);
    v8f acc = {};
    acc = __builtin_amdgcn_wmma_f32_16x16x32_f16(
        false, A, false, B, (short)0, acc, false, false);
    const int ncol = wave * 16 + (lane & 15);
    if (ncol < 40) {
      const long mbase = (long)blockIdx.x * 16 + hi * 8;
#pragma unroll
      for (int rr = 0; rr < 8; ++rr)
        out[(mbase + rr) * 40 + ncol] = acc[rr];
    }
  }
}

// ---------------------------------------------------------------------------

extern "C" void kernel_launch(void* const* d_in, const int* in_sizes, int n_in,
                              void* d_out, int out_size, void* d_ws,
                              size_t ws_size, hipStream_t stream) {
  const float* feat  = (const float*)d_in[0];  // [100000, 512]
  const float* W1    = (const float*)d_in[1];  // [32, 1024]
  const float* W2    = (const float*)d_in[2];  // [32, 64]
  const float* Wc    = (const float*)d_in[3];  // [40, 32]
  const int* nodes   = (const int*)d_in[4];    // [16384]
  const int* neigh1  = (const int*)d_in[5];    // [100000, 5]
  const int* neigh2  = (const int*)d_in[6];    // [16384, 5]
  float* out = (float*)d_out;                  // [16384, 40]

  char* ws = (char*)d_ws;
  float* h1 = (float*)ws;  // [98304, 32] f32 = 12,582,912 B
  size_t off = (size_t)98304 * 32 * sizeof(float);
  unsigned short* w1pk = (unsigned short*)(ws + off);
  off += (size_t)2 * 32 * 32 * 16 * 2;  // 64 KB
  unsigned short* w2pk = (unsigned short*)(ws + off);
  off += (size_t)2 * 2 * 32 * 16 * 2;   // 4 KB
  unsigned short* wcpk = (unsigned short*)(ws + off);
  off += (size_t)3 * 32 * 16 * 2;       // 3 KB

  pack_w1_kernel<<<128, 256, 0, stream>>>(W1, w1pk);
  pack_w2_kernel<<<8, 256, 0, stream>>>(W2, w2pk);
  pack_wc_kernel<<<6, 256, 0, stream>>>(Wc, wcpk);

  enc1_kernel<<<98304 / 16, 256, 0, stream>>>(feat, nodes, neigh1, neigh2,
                                              w1pk, h1);
  layer2_kernel<<<16384 / 16, 256, 0, stream>>>(h1, w2pk, wcpk, out);
}
